// MultiHeadAttention_69930657513852
// MI455X (gfx1250) — matrix operand order repaired and checked by reading, not compiled, not run on hardware
//
#include <hip/hip_runtime.h>
#include <math.h>
#include <stdint.h>

typedef __attribute__((ext_vector_type(16))) __bf16 v16bf;
typedef __attribute__((ext_vector_type(8)))  __bf16 v8bf;
typedef __attribute__((ext_vector_type(4)))  __bf16 v4bf;
typedef __attribute__((ext_vector_type(8)))  float  v8f;
typedef __attribute__((ext_vector_type(4)))  float  v4f;

#define BVAL  2
#define LVAL  2048
#define DVAL  1024
#define HVAL  16
#define HDVAL 64

// ------------------------------------------------------- async global->LDS
#if defined(__has_builtin)
#if __has_builtin(__builtin_amdgcn_global_load_async_to_lds_b128)
#define HAVE_ASYNC_LDS 1
#endif
#endif

#ifdef HAVE_ASYNC_LDS
// builtin expects pointers to gcc-vector int4 in explicit address spaces
typedef int gint4 __attribute__((vector_size(16)));
typedef __attribute__((address_space(1))) gint4 ga_int4;  // global
typedef __attribute__((address_space(3))) gint4 ls_int4;  // LDS
__device__ __forceinline__ void async_copy_b128(const void* g, void* l) {
  // generic LDS pointer low 32 bits == LDS offset (ISA 10.2 aperture rules)
  __builtin_amdgcn_global_load_async_to_lds_b128(
      (ga_int4*)(uintptr_t)g, (ls_int4*)(uint32_t)(uintptr_t)l, 0, 0);
}
__device__ __forceinline__ void wait_async() {
#if __has_builtin(__builtin_amdgcn_s_wait_asynccnt)
  __builtin_amdgcn_s_wait_asynccnt(0);
#else
  asm volatile("s_wait_asynccnt 0x0" ::: "memory");
#endif
}
#else
__device__ __forceinline__ void async_copy_b128(const void* g, void* l) {
  *(v8bf*)l = *(const v8bf*)g;
}
__device__ __forceinline__ void wait_async() {}
#endif

// ---------------------------------------------------------------- conversion
__global__ void f32_to_bf16_v4(const float* __restrict__ in,
                               __bf16* __restrict__ out, int n4) {
  int i = blockIdx.x * blockDim.x + threadIdx.x;
  if (i < n4) {
    float4 v = ((const float4*)in)[i];
    v4bf o;
    o[0] = (__bf16)v.x; o[1] = (__bf16)v.y; o[2] = (__bf16)v.z; o[3] = (__bf16)v.w;
    ((v4bf*)out)[i] = o;
  }
}

// ---------------------------------------------------------------- GEMM
// C[M,1024] = A[M,1024] * B[1024,1024], bf16 in, f32 accumulate.
// 256 threads = 8 waves, tile 128x64, BK=32; waves 4x2, each 32x32 (2x2 WMMA).
template <int STORE_BF16>
__global__ __launch_bounds__(256)
void gemm_wmma(const __bf16* __restrict__ A, const __bf16* __restrict__ Bm,
               float* __restrict__ Cf, __bf16* __restrict__ Cb) {
  constexpr int N = 1024;
  constexpr int K = 1024;
  __shared__ union {
    struct { __bf16 a[128 * 32]; __bf16 b[32 * 64]; } ab;
    __bf16 cb[128 * 64];
    float  cf[128 * 64];
  } sm;

  const int tid  = threadIdx.x;
  const int lane = tid & 31;
  const int wave = tid >> 5;
  const int wr   = wave >> 1;   // 0..3
  const int wc   = wave & 1;    // 0..1
  const int hb   = lane >> 4;
  const int ln   = lane & 15;
  const int row0 = blockIdx.y * 128;
  const int col0 = blockIdx.x * 64;

  v8f acc[2][2];
  v8f zero = {};
  #pragma unroll
  for (int mi = 0; mi < 2; ++mi)
    #pragma unroll
    for (int ni = 0; ni < 2; ++ni)
      acc[mi][ni] = zero;

  for (int k0 = 0; k0 < K; k0 += 32) {
    __syncthreads();
    {  // B tile 32x64
      int lin = tid * 8;
      int r = lin >> 6, c = lin & 63;
      async_copy_b128(&Bm[(size_t)(k0 + r) * N + col0 + c], &sm.ab.b[lin]);
    }
    #pragma unroll
    for (int p = 0; p < 2; ++p) {  // A tile 128x32
      int lin = (p * 256 + tid) * 8;
      int r = lin >> 5, c = lin & 31;
      async_copy_b128(&A[(size_t)(row0 + r) * K + k0 + c], &sm.ab.a[lin]);
    }
    wait_async();
    __syncthreads();

    v16bf af[2], bfr[2];
    #pragma unroll
    for (int mi = 0; mi < 2; ++mi) {
      int rl = wr * 32 + mi * 16 + ln;
      #pragma unroll
      for (int e = 0; e < 16; ++e) {
        int k = (e < 8) ? (hb * 8 + e) : (16 + hb * 8 + (e - 8));
        af[mi][e] = sm.ab.a[rl * 32 + k];
      }
    }
    #pragma unroll
    for (int ni = 0; ni < 2; ++ni) {
      int cl = wc * 32 + ni * 16 + ln;
      #pragma unroll
      for (int e = 0; e < 16; ++e) {
        int k = (e < 8) ? (hb * 8 + e) : (16 + hb * 8 + (e - 8));
        bfr[ni][e] = sm.ab.b[k * 64 + cl];
      }
    }
    #pragma unroll
    for (int mi = 0; mi < 2; ++mi)
      #pragma unroll
      for (int ni = 0; ni < 2; ++ni)
        acc[mi][ni] = __builtin_amdgcn_wmma_f32_16x16x32_bf16(
            false, af[mi], false, bfr[ni], (short)0, acc[mi][ni], false, false);
  }

  // ---- epilogue: stage C tile in LDS, then coalesced vector stores
  __syncthreads();
  if constexpr (STORE_BF16) {
    #pragma unroll
    for (int mi = 0; mi < 2; ++mi)
      #pragma unroll
      for (int ni = 0; ni < 2; ++ni)
        #pragma unroll
        for (int i = 0; i < 8; ++i)
          sm.cb[(wr * 32 + mi * 16 + i + hb * 8) * 64 + wc * 32 + ni * 16 + ln] =
              (__bf16)acc[mi][ni][i];
    __syncthreads();
    #pragma unroll
    for (int p = 0; p < 4; ++p) {
      int lin = (p * 256 + tid) * 8;
      int r = lin >> 6, c = lin & 63;
      *(v8bf*)&Cb[(size_t)(row0 + r) * N + col0 + c] = *(v8bf*)&sm.cb[lin];
    }
  } else {
    #pragma unroll
    for (int mi = 0; mi < 2; ++mi)
      #pragma unroll
      for (int ni = 0; ni < 2; ++ni)
        #pragma unroll
        for (int i = 0; i < 8; ++i)
          sm.cf[(wr * 32 + mi * 16 + i + hb * 8) * 64 + wc * 32 + ni * 16 + ln] =
              acc[mi][ni][i];
    __syncthreads();
    #pragma unroll
    for (int p = 0; p < 8; ++p) {
      int lin = (p * 256 + tid) * 4;
      int r = lin >> 6, c = lin & 63;
      *(v4f*)&Cf[(size_t)(row0 + r) * N + col0 + c] = *(v4f*)&sm.cf[lin];
    }
  }
}

// ---------------------------------------------------------------- attention
// grid: (L/64, B*H), block 128 = 4 waves; wave w owns 16 query rows.
__global__ __launch_bounds__(128)
void attn_wmma(const __bf16* __restrict__ Q, const __bf16* __restrict__ Kb,
               const __bf16* __restrict__ V, __bf16* __restrict__ O) {
  __shared__ __bf16 Ks[64 * 64];
  __shared__ __bf16 Vs[64 * 64];
  __shared__ __bf16 Ps[4 * 16 * 64];  // per-wave P staging (C->A relayout)

  const int tid  = threadIdx.x;
  const int lane = tid & 31;
  const int w    = tid >> 5;
  const int hb   = lane >> 4;
  const int ln   = lane & 15;
  const int bh   = blockIdx.y;
  const int b    = bh >> 4;  // H = 16
  const int h    = bh & 15;
  const int qblock = blockIdx.x;
  const int qrow0  = qblock * 64;
  const size_t headoff = (size_t)h * HDVAL;

  // Q A-fragments (16 rows x 64 hd = two 16x32 frags), loaded once
  v16bf qa[2];
  {
    int row = qrow0 + w * 16 + ln;
    const __bf16* qp = Q + (size_t)(b * LVAL + row) * DVAL + headoff;
    #pragma unroll
    for (int s = 0; s < 2; ++s)
      #pragma unroll
      for (int e = 0; e < 16; ++e) {
        int k = (e < 8) ? (hb * 8 + e) : (16 + hb * 8 + (e - 8));
        qa[s][e] = qp[s * 32 + k];
      }
  }

  float m[8], l[8];
  v8f o[4];
  v8f zero = {};
  #pragma unroll
  for (int i = 0; i < 8; ++i) { m[i] = -1e30f; l[i] = 0.0f; }
  #pragma unroll
  for (int f = 0; f < 4; ++f) o[f] = zero;

  for (int jb = 0; jb <= qblock; ++jb) {
    __syncthreads();
    #pragma unroll
    for (int p = 0; p < 4; ++p) {  // K/V 64x64 tiles
      int lin = (p * 128 + tid) * 8;
      int r = lin >> 6, c = lin & 63;
      size_t g = (size_t)(b * LVAL + jb * 64 + r) * DVAL + headoff + c;
      async_copy_b128(&Kb[g], &Ks[lin]);
      async_copy_b128(&V[g],  &Vs[lin]);
    }
    wait_async();
    __syncthreads();

    // S = Q * K^T (16 rows x 64 keys per wave)
    v8f sacc[4];
    #pragma unroll
    for (int f = 0; f < 4; ++f) sacc[f] = zero;
    #pragma unroll
    for (int s = 0; s < 2; ++s) {
      #pragma unroll
      for (int f = 0; f < 4; ++f) {
        v16bf kf;
        #pragma unroll
        for (int e = 0; e < 16; ++e) {
          int k = (e < 8) ? (hb * 8 + e) : (16 + hb * 8 + (e - 8));
          kf[e] = Ks[(f * 16 + ln) * 64 + s * 32 + k];  // B[k][n]=K[key n][hd k]
        }
        sacc[f] = __builtin_amdgcn_wmma_f32_16x16x32_bf16(
            false, qa[s], false, kf, (short)0, sacc[f], false, false);
      }
    }

    // scale + causal mask (diagonal block only)
    const float sc = 0.125f;  // 1/sqrt(64)
    #pragma unroll
    for (int f = 0; f < 4; ++f)
      #pragma unroll
      for (int i = 0; i < 8; ++i) {
        float v = sacc[f][i] * sc;
        if (jb == qblock) {
          int rr = w * 16 + i + hb * 8;
          int cc = f * 16 + ln;
          if (cc > rr) v = -1e30f;
        }
        sacc[f][i] = v;
      }

    // online softmax: each C VGPR's rows live in 16-lane halves
    #pragma unroll
    for (int i = 0; i < 8; ++i) {
      float rm = sacc[0][i];
      #pragma unroll
      for (int f = 1; f < 4; ++f) rm = fmaxf(rm, sacc[f][i]);
      #pragma unroll
      for (int off = 1; off < 16; off <<= 1) rm = fmaxf(rm, __shfl_xor(rm, off, 32));
      float nm = fmaxf(m[i], rm);
      float alpha = __expf(m[i] - nm);
      float rs = 0.0f;
      #pragma unroll
      for (int f = 0; f < 4; ++f) {
        float p = __expf(sacc[f][i] - nm);
        sacc[f][i] = p;
        rs += p;
      }
      #pragma unroll
      for (int off = 1; off < 16; off <<= 1) rs += __shfl_xor(rs, off, 32);
      l[i] = l[i] * alpha + rs;
      m[i] = nm;
      #pragma unroll
      for (int f = 0; f < 4; ++f) o[f][i] *= alpha;
    }

    // stage P through LDS: C-layout -> A-layout
    #pragma unroll
    for (int f = 0; f < 4; ++f)
      #pragma unroll
      for (int i = 0; i < 8; ++i)
        Ps[w * 1024 + (i + hb * 8) * 64 + f * 16 + ln] = (__bf16)sacc[f][i];
    __syncthreads();

    // O += P * V
    #pragma unroll
    for (int s = 0; s < 2; ++s) {
      v16bf pa;
      #pragma unroll
      for (int e = 0; e < 16; ++e) {
        int k = (e < 8) ? (hb * 8 + e) : (16 + hb * 8 + (e - 8));
        pa[e] = Ps[w * 1024 + ln * 64 + s * 32 + k];
      }
      #pragma unroll
      for (int f = 0; f < 4; ++f) {
        v16bf vf;
        #pragma unroll
        for (int e = 0; e < 16; ++e) {
          int k = (e < 8) ? (hb * 8 + e) : (16 + hb * 8 + (e - 8));
          vf[e] = Vs[(s * 32 + k) * 64 + f * 16 + ln];  // B[k][n]=V[key k][hd n]
        }
        o[f] = __builtin_amdgcn_wmma_f32_16x16x32_bf16(
            false, pa, false, vf, (short)0, o[f], false, false);
      }
    }
  }

  // normalize, store bf16 [B, L, H*HD]
  #pragma unroll
  for (int f = 0; f < 4; ++f)
    #pragma unroll
    for (int i = 0; i < 8; ++i) {
      int row = qrow0 + w * 16 + i + hb * 8;
      float v = o[f][i] / l[i];
      O[(size_t)(b * LVAL + row) * DVAL + headoff + f * 16 + ln] = (__bf16)v;
    }
}

// ---------------------------------------------------------------- launch
extern "C" void kernel_launch(void* const* d_in, const int* in_sizes, int n_in,
                              void* d_out, int out_size, void* d_ws, size_t ws_size,
                              hipStream_t stream) {
  (void)in_sizes; (void)n_in; (void)out_size; (void)ws_size;
  // setup_inputs order: x, Wk, Wq, Wv, Wo
  const float* x  = (const float*)d_in[0];
  const float* Wk = (const float*)d_in[1];
  const float* Wq = (const float*)d_in[2];
  const float* Wv = (const float*)d_in[3];
  const float* Wo = (const float*)d_in[4];
  float* out = (float*)d_out;

  const int M  = BVAL * LVAL;  // 4096
  const int Dm = DVAL;         // 1024
  const size_t nx = (size_t)M * Dm;   // 4M elems
  const size_t nw = (size_t)Dm * Dm;  // 1M elems

  __bf16* xb  = (__bf16*)d_ws;
  __bf16* wqb = xb  + nx;
  __bf16* wkb = wqb + nw;
  __bf16* wvb = wkb + nw;
  __bf16* wob = wvb + nw;
  __bf16* qb  = wob + nw;
  __bf16* kb  = qb  + nx;
  __bf16* vb  = kb  + nx;
  __bf16* ab  = vb  + nx;

  f32_to_bf16_v4<<<(int)((nx / 4 + 255) / 256), 256, 0, stream>>>(x,  xb,  (int)(nx / 4));
  f32_to_bf16_v4<<<(int)((nw / 4 + 255) / 256), 256, 0, stream>>>(Wq, wqb, (int)(nw / 4));
  f32_to_bf16_v4<<<(int)((nw / 4 + 255) / 256), 256, 0, stream>>>(Wk, wkb, (int)(nw / 4));
  f32_to_bf16_v4<<<(int)((nw / 4 + 255) / 256), 256, 0, stream>>>(Wv, wvb, (int)(nw / 4));
  f32_to_bf16_v4<<<(int)((nw / 4 + 255) / 256), 256, 0, stream>>>(Wo, wob, (int)(nw / 4));

  dim3 gg(Dm / 64, M / 128);
  gemm_wmma<1><<<gg, 256, 0, stream>>>(xb, wqb, nullptr, qb);
  gemm_wmma<1><<<gg, 256, 0, stream>>>(xb, wkb, nullptr, kb);
  gemm_wmma<1><<<gg, 256, 0, stream>>>(xb, wvb, nullptr, vb);

  dim3 ga(LVAL / 64, BVAL * HVAL);
  attn_wmma<<<ga, 128, 0, stream>>>(qb, kb, vb, ab);

  gemm_wmma<0><<<gg, 256, 0, stream>>>(ab, wob, out, nullptr);
}
